// SparseCNN_50311246905735
// MI455X (gfx1250) — matrix operand order repaired and checked by reading, not compiled, not run on hardware
//
#include <hip/hip_runtime.h>
#include <hip/hip_bf16.h>

typedef __attribute__((ext_vector_type(16))) __bf16 v16bf;
typedef __attribute__((ext_vector_type(8)))  float  v8f;
typedef __attribute__((ext_vector_type(4)))  unsigned v4u;
typedef __attribute__((ext_vector_type(8)))  unsigned v8u;

#define BATCH 8192

// ---------------------------------------------------------------------------
// Utility: zero a float buffer (graph-capture-safe replacement for memset)
// ---------------------------------------------------------------------------
__global__ void zero_f32_kernel(float* p, int n) {
    int i = blockIdx.x * blockDim.x + threadIdx.x;
    if (i < n) p[i] = 0.0f;
}

// ---------------------------------------------------------------------------
// One-shot f32 -> bf16 weight conversion (weights then TDM-copied per block)
// ---------------------------------------------------------------------------
__global__ void cvt_f32_bf16_kernel(const float* __restrict__ src,
                                    __hip_bfloat16* __restrict__ dst, int n) {
    int i = blockIdx.x * blockDim.x + threadIdx.x;
    if (i < n) dst[i] = __float2bfloat16(src[i]);
}

// ---------------------------------------------------------------------------
// Conv0: 1 -> 32 channels, 3x3, stride 1, SAME padding.  Direct conv (K=9 is
// too skinny for WMMA).  Writes pre-BN y0 (bias added) as bf16 and
// accumulates per-channel sum / sum-of-squares for BatchNorm.
// One block per image; 30x30 zero-padded tile staged in LDS.
// ---------------------------------------------------------------------------
__global__ __launch_bounds__(256) void conv0_3x3_kernel(
    const float* __restrict__ x,      // (B,1,28,28)
    const float* __restrict__ w,      // (32,1,3,3)
    const float* __restrict__ bias,   // (32)
    __hip_bfloat16* __restrict__ y0,  // (B,32,28,28) pre-BN
    float* __restrict__ gsum, float* __restrict__ gsq)
{
    __shared__ float xs[30 * 30];
    __shared__ float ws[32 * 9];
    __shared__ float wb[32];
    __shared__ float ssum[32], ssq[32];

    const int t = threadIdx.x;
    const int b = blockIdx.x;

    for (int i = t; i < 900; i += 256) xs[i] = 0.0f;
    for (int i = t; i < 288; i += 256) ws[i] = w[i];
    if (t < 32) { wb[t] = bias[t]; ssum[t] = 0.0f; ssq[t] = 0.0f; }
    __syncthreads();

    const float* xb = x + (size_t)b * 784;
    for (int i = t; i < 784; i += 256) {
        int py = i / 28, px = i % 28;
        xs[(py + 1) * 30 + (px + 1)] = xb[i];
    }
    __syncthreads();

    for (int oc = 0; oc < 32; ++oc) {
        float s = 0.0f, sq = 0.0f;
        for (int p = t; p < 784; p += 256) {
            int py = p / 28, px = p % 28;
            float acc = wb[oc];
#pragma unroll
            for (int ky = 0; ky < 3; ++ky)
#pragma unroll
                for (int kx = 0; kx < 3; ++kx)
                    acc += xs[(py + ky) * 30 + (px + kx)] * ws[oc * 9 + ky * 3 + kx];
            y0[((size_t)b * 32 + oc) * 784 + p] = __float2bfloat16(acc);
            s += acc;
            sq += acc * acc;
        }
        atomicAdd(&ssum[oc], s);
        atomicAdd(&ssq[oc], sq);
    }
    __syncthreads();
    if (t < 32) {
        atomicAdd(&gsum[t], ssum[t]);
        atomicAdd(&gsq[t], ssq[t]);
    }
}

// ---------------------------------------------------------------------------
// BN finalize: per-channel scale/shift from accumulated sums.
//   norm(y) = y*scale + shift ; scale = g*rsqrt(var+eps), shift = be - mean*scale
// ---------------------------------------------------------------------------
__global__ void bn_finalize_kernel(const float* __restrict__ gsum,
                                   const float* __restrict__ gsq,
                                   const float* __restrict__ g,
                                   const float* __restrict__ be,
                                   float* __restrict__ scale,
                                   float* __restrict__ shift,
                                   int C, float inv_count)
{
    int c = blockIdx.x * blockDim.x + threadIdx.x;
    if (c >= C) return;
    float mean = gsum[c] * inv_count;
    float var  = gsq[c] * inv_count - mean * mean;
    float sc   = g[c] * rsqrtf(var + 1e-5f);
    scale[c] = sc;
    shift[c] = be[c] - mean * sc;
}

// ---------------------------------------------------------------------------
// Stride-2 2x2 conv as a WMMA GEMM:
//   C[oc][n] = sum_k W[oc][k] * P[k][n],  k = ic*4 + ky*2 + kx  (== OIHW flat)
//   n = b*OH*OW + oy*OW + ox, input pixel (2oy+ky, 2ox+kx)
// Input activations are pre-BN bf16; BN scale/shift + relu applied on load.
// Block: 256 threads = 8 waves laid out 4(M) x 2(N); tile M=64, N=32.
// A (weight) tile: bf16, contiguous (OC,K) row-major run -> fetched into LDS
// by the Tensor Data Mover (1D descriptor), overlapped with B staging, fenced
// with s_wait_tensorcnt.  Each wave: K/32 v_wmma_f32_16x16x32_bf16 steps.
// Epilogue adds bias, writes pre-BN bf16 output, accumulates BN stats.
// ---------------------------------------------------------------------------
template <int IC, int OC, int IH, int IW, int OH, int OW>
__global__ __launch_bounds__(256) void conv_down2_wmma_kernel(
    const __hip_bfloat16* __restrict__ yin,   // (B,IC,IH,IW) pre-BN
    const __hip_bfloat16* __restrict__ wbf,   // (OC,K) bf16 weights
    const float* __restrict__ bias,           // (OC)
    const float* __restrict__ scale_in,       // (IC) BN for input
    const float* __restrict__ shift_in,       // (IC)
    __hip_bfloat16* __restrict__ yout,        // (B,OC,OH,OW) pre-BN
    float* __restrict__ gsum, float* __restrict__ gsq)
{
    constexpr int K  = IC * 4;
    constexpr int NT = 32;   // N tile (block)
    constexpr int MT = 64;   // M tile (block)

    __shared__ __hip_bfloat16 lA[MT * K];   // weights  [m][k]   (TDM-filled)
    __shared__ __hip_bfloat16 lB[NT * K];   // activations [n][k]
    __shared__ float ssum[MT], ssq[MT];

    const int t   = threadIdx.x;
    const int n0  = blockIdx.x * NT;
    const int oc0 = blockIdx.y * MT;

    if (t < MT) { ssum[t] = 0.0f; ssq[t] = 0.0f; }

    // --- stage A via Tensor Data Mover: contiguous MT*K bf16 run -> LDS ---
    // D# group0: count=1 | lds_addr | 57-bit global_addr | type=2 ("image")
    // D# group1: data_size=2B, tensor_dim0 = tile_dim0 = MT*K (1D tile),
    //            tensor_dim1 = 1, dims 1/2 of tile unused, stride0 = MT*K.
    if (t < 32) {   // wave 0 only (TDM ignores EXEC; issue once per block)
        constexpr unsigned NE = (unsigned)(MT * K);   // 8192 / 16384 elems
        unsigned ldsoff = (unsigned)(size_t)&lA[0];   // low 32 bits = LDS offset
        unsigned long long ga =
            (unsigned long long)(size_t)(wbf + (size_t)oc0 * K);
        v4u g0;
        g0[0] = 1u;                                        // count=1, user mode
        g0[1] = ldsoff;                                    // lds_addr
        g0[2] = (unsigned)ga;                              // global_addr[31:0]
        g0[3] = ((unsigned)(ga >> 32) & 0x01ffffffu) | (2u << 30); // [56:32]|type
        v8u g1;
        g1[0] = 1u << 16;                                  // data_size = 2 bytes
        g1[1] = (NE & 0xffffu) << 16;                      // tensor_dim0[15:0]
        g1[2] = (NE >> 16) | (1u << 16);                   // dim0[31:16] | dim1=1
        g1[3] = (NE & 0xffffu) << 16;                      // tile_dim0 = NE
        g1[4] = 0u;                                        // tile_dim1/2 unused
        g1[5] = NE;                                        // tensor_dim0_stride
        g1[6] = 0u;
        g1[7] = 0u;
        asm volatile("tensor_load_to_lds %0, %1" :: "s"(g0), "s"(g1) : "memory");
    }

    // --- stage B: input patches with BN + relu applied, NT x K ---
    {
        const int nl  = t >> 3;                 // 0..31
        const int n   = n0 + nl;
        const int bb  = n / (OH * OW);
        const int rem = n % (OH * OW);
        const int oy  = rem / OW, ox = rem % OW;
        constexpr int KPT = K / 8;              // k-values per thread
        const int kbase = (t & 7) * KPT;
#pragma unroll 4
        for (int kk = 0; kk < KPT; ++kk) {
            int k  = kbase + kk;
            int ic = k >> 2, ky = (k >> 1) & 1, kx = k & 1;
            float v = __bfloat162float(
                yin[((size_t)(bb * IC + ic) * IH + (2 * oy + ky)) * IW + (2 * ox + kx)]);
            v = v * scale_in[ic] + shift_in[ic];
            v = v > 0.0f ? v : 0.0f;
            lB[nl * K + k] = __float2bfloat16(v);
        }
    }

    // Fence the TDM copy (wave 0), then block-wide barrier orders everyone.
    if (t < 32) __builtin_amdgcn_s_wait_tensorcnt(0);
    __syncthreads();

    const int lane = t & 31;
    const int wave = t >> 5;
    const int wm   = wave >> 1;        // 0..3
    const int wn   = wave & 1;         // 0..1
    const int l15  = lane & 15;
    const int lhi  = lane >> 4;        // 0 or 1

    const int arow = wm * 16 + l15;    // A row (M) held by this lane
    const int bcol = wn * 16 + l15;    // B col (N) held by this lane

    v8f c = {};
    for (int s = 0; s < K; s += 32) {
        union { v16bf v; unsigned u[8]; } af, bf;
#pragma unroll
        for (int q = 0; q < 8; ++q) {
            // A 16x32 bf16 layout: vgpr q holds K pair at
            //   (q<4 ? 0 : 16) + (lane>=16 ? 8 : 0) + (q&3)*2
            int ka = s + ((q >= 4) ? 16 : 0) + (lhi ? 8 : 0) + (q & 3) * 2;
            af.u[q] = *(const unsigned*)&lA[arow * K + ka];
            // B 32x16 bf16 layout: vgpr q holds K pair at
            //   (q<4 ? 0 : 16) + (lane>=16 ? 8 : 0) + (q&3)*2
            int kb = s + ((q >= 4) ? 16 : 0) + (lhi ? 8 : 0) + (q & 3) * 2;
            bf.u[q] = *(const unsigned*)&lB[bcol * K + kb];
        }
        c = __builtin_amdgcn_wmma_f32_16x16x32_bf16(
                false, af.v, false, bf.v, (short)0, c, false, false);
    }

    // --- epilogue: bias, store pre-BN bf16, BN stats ---
    const int n   = n0 + bcol;
    const int bb  = n / (OH * OW);
    const int rem = n % (OH * OW);
#pragma unroll
    for (int r = 0; r < 8; ++r) {
        int m  = wm * 16 + lhi * 8 + r;   // C/D layout: vgpr r, lanes16-31 -> M+8
        int oc = oc0 + m;
        float v = c[r] + bias[oc];
        yout[(size_t)(bb * OC + oc) * (OH * OW) + rem] = __float2bfloat16(v);
        atomicAdd(&ssum[m], v);
        atomicAdd(&ssq[m], v * v);
    }
    __syncthreads();
    if (t < MT) {
        atomicAdd(&gsum[oc0 + t], ssum[t]);
        atomicAdd(&gsq[oc0 + t], ssq[t]);
    }
}

// ---------------------------------------------------------------------------
// BN2 + relu + global average pool + FC(128->10).  One block per image.
// ---------------------------------------------------------------------------
__global__ __launch_bounds__(128) void pool_fc_kernel(
    const __hip_bfloat16* __restrict__ y2,    // (B,128,7,7) pre-BN
    const float* __restrict__ scale2,
    const float* __restrict__ shift2,
    const float* __restrict__ wfc,            // (10,128)
    const float* __restrict__ bfc,            // (10)
    float* __restrict__ out)                  // (B,10)
{
    __shared__ float pooled[128];
    const int b = blockIdx.x;
    const int c = threadIdx.x;

    const float sc = scale2[c], sh = shift2[c];
    const __hip_bfloat16* p = y2 + ((size_t)b * 128 + c) * 49;
    float acc = 0.0f;
#pragma unroll
    for (int s = 0; s < 49; ++s) {
        float v = __bfloat162float(p[s]) * sc + sh;
        acc += v > 0.0f ? v : 0.0f;
    }
    pooled[c] = acc * (1.0f / 49.0f);
    __syncthreads();

    if (c < 10) {
        float a = bfc[c];
#pragma unroll 8
        for (int k = 0; k < 128; ++k) a += pooled[k] * wfc[c * 128 + k];
        out[(size_t)b * 10 + c] = a;
    }
}

// ---------------------------------------------------------------------------
// Launch
// ---------------------------------------------------------------------------
extern "C" void kernel_launch(void* const* d_in, const int* in_sizes, int n_in,
                              void* d_out, int out_size, void* d_ws, size_t ws_size,
                              hipStream_t stream) {
    const float* x    = (const float*)d_in[0];
    const float* w0   = (const float*)d_in[1];
    const float* b0   = (const float*)d_in[2];
    const float* g0   = (const float*)d_in[3];
    const float* be0  = (const float*)d_in[4];
    const float* w1   = (const float*)d_in[5];
    const float* b1   = (const float*)d_in[6];
    const float* g1   = (const float*)d_in[7];
    const float* be1  = (const float*)d_in[8];
    const float* w2   = (const float*)d_in[9];
    const float* b2   = (const float*)d_in[10];
    const float* g2   = (const float*)d_in[11];
    const float* be2  = (const float*)d_in[12];
    const float* wfc  = (const float*)d_in[13];
    const float* bfc  = (const float*)d_in[14];
    float* out = (float*)d_out;

    // Workspace: [stats 4KB | wb1 bf16 16KB | wb2 bf16 64KB | pad | y0 | y1 | y2]
    float* stats = (float*)d_ws;
    float* sum0   = stats + 0;    // 32
    float* sq0    = stats + 32;   // 32
    float* sum1   = stats + 64;   // 64
    float* sq1    = stats + 128;  // 64
    float* sum2   = stats + 192;  // 128
    float* sq2    = stats + 320;  // 128
    float* scale0 = stats + 448;  // 32
    float* shift0 = stats + 480;  // 32
    float* scale1 = stats + 512;  // 64
    float* shift1 = stats + 576;  // 64
    float* scale2 = stats + 640;  // 128
    float* shift2 = stats + 768;  // 128

    __hip_bfloat16* wb1 = (__hip_bfloat16*)((char*)d_ws + 4096);   // 64*128
    __hip_bfloat16* wb2 = wb1 + 64 * 128;                          // 128*256

    char* base = (char*)d_ws + 131072;
    const size_t Y0 = (size_t)BATCH * 32 * 784;   // 205,520,896 elems
    const size_t Y1 = (size_t)BATCH * 64 * 196;   // 102,760,448
    const size_t Y2 = (size_t)BATCH * 128 * 49;   //  51,380,224
    __hip_bfloat16* y0 = (__hip_bfloat16*)base;
    __hip_bfloat16* y1 = y0 + Y0;
    __hip_bfloat16* y2 = y1 + Y1;

    // Zero stat accumulators + pre-convert GEMM weights to bf16
    zero_f32_kernel<<<2, 256, 0, stream>>>(stats, 448);
    cvt_f32_bf16_kernel<<<32, 256, 0, stream>>>(w1, wb1, 64 * 128);
    cvt_f32_bf16_kernel<<<128, 256, 0, stream>>>(w2, wb2, 128 * 256);

    // Layer 0: direct conv + stats
    conv0_3x3_kernel<<<BATCH, 256, 0, stream>>>(x, w0, b0, y0, sum0, sq0);
    bn_finalize_kernel<<<1, 32, 0, stream>>>(sum0, sq0, g0, be0, scale0, shift0,
                                             32, 1.0f / ((float)BATCH * 784.0f));

    // Layer 1: WMMA GEMM, M=64, K=128, N = 8192*196 = 1,605,632 (= 50176 * 32)
    conv_down2_wmma_kernel<32, 64, 28, 28, 14, 14>
        <<<dim3(50176, 1), 256, 0, stream>>>(y0, wb1, b1, scale0, shift0, y1, sum1, sq1);
    bn_finalize_kernel<<<1, 64, 0, stream>>>(sum1, sq1, g1, be1, scale1, shift1,
                                             64, 1.0f / ((float)BATCH * 196.0f));

    // Layer 2: WMMA GEMM, M=128, K=256, N = 8192*49 = 401,408 (= 12544 * 32)
    conv_down2_wmma_kernel<64, 128, 14, 14, 7, 7>
        <<<dim3(12544, 2), 256, 0, stream>>>(y1, wb2, b2, scale1, shift1, y2, sum2, sq2);
    bn_finalize_kernel<<<1, 128, 0, stream>>>(sum2, sq2, g2, be2, scale2, shift2,
                                              128, 1.0f / ((float)BATCH * 49.0f));

    // BN2 + relu + pool + FC
    pool_fc_kernel<<<BATCH, 128, 0, stream>>>(y2, scale2, shift2, wfc, bfc, out);
}
